// MultiHeadAttention_35493609734535
// MI455X (gfx1250) — matrix operand order repaired and checked by reading, not compiled, not run on hardware
//
#include <hip/hip_runtime.h>
#include <hip/hip_bf16.h>

// ---------------------------------------------------------------------------
// GPT-2 multi-head attention block for MI455X (gfx1250).
// f16 WMMA / f32 accumulate; TDM (tensor_load_to_lds) streams weight tiles
// into LDS with TENSORcnt double-buffering for the two big GEMMs.
// ---------------------------------------------------------------------------

#define B_ 2
#define T_ 2048
#define C_ 1024
#define H_ 16
#define D_ 64
#define BT_ (B_ * T_)      // 4096
#define NQKV_ (3 * C_)     // 3072

typedef __attribute__((ext_vector_type(16))) _Float16 v16h;
typedef __attribute__((ext_vector_type(8)))  _Float16 v8h;
typedef __attribute__((ext_vector_type(8)))  float    v8f;
typedef __attribute__((ext_vector_type(4)))  unsigned int u32x4_t;
typedef __attribute__((ext_vector_type(4)))  int      i32x4_t;
typedef __attribute__((ext_vector_type(8)))  int      i32x8_t;

#define NEG_INF (-3.0e38f)

#if defined(__has_builtin)
#  if __has_builtin(__builtin_amdgcn_tensor_load_to_lds)
#    define HAVE_TDM 1
#  else
#    define HAVE_TDM 0
#  endif
#else
#  define HAVE_TDM 0
#endif

// ---- TENSORcnt waits (probe-confirmed builtin; asm fallback) ---------------
static __device__ __forceinline__ void wait_tensorcnt0() {
#if __has_builtin(__builtin_amdgcn_s_wait_tensorcnt)
  __builtin_amdgcn_s_wait_tensorcnt(0);
#else
  asm volatile("s_wait_tensorcnt 0x0" ::: "memory");
#endif
}
static __device__ __forceinline__ void wait_tensorcnt1() {
#if __has_builtin(__builtin_amdgcn_s_wait_tensorcnt)
  __builtin_amdgcn_s_wait_tensorcnt(1);
#else
  asm volatile("s_wait_tensorcnt 0x1" ::: "memory");
#endif
}

// ---- TDM: async-load a 64(rows) x 32(cols) f16 tile into LDS ---------------
// Source is row-major [Nrows][1024] f16; tile start at gsrc; dest ldsdst.
// D# per CDNA5 ISA ch.8: group0 {count=1, lds_addr, global_addr, type=2},
// group1 {data_size=2B, tensor_dim0=1024, tensor_dim1=Nrows, tile=32x64,
// tensor_dim0_stride=1024}. 2D tensor -> groups 2/3 zero.
static __device__ __forceinline__ void tdm_issue_64x32(const _Float16* gsrc,
                                                       void* ldsdst,
                                                       int Nrows) {
#if HAVE_TDM
  unsigned long long ga = (unsigned long long)(uintptr_t)gsrc;
  unsigned lds_off = (unsigned)(uintptr_t)ldsdst;   // low 32 bits = LDS byte addr
  u32x4_t g0;
  g0[0] = 1u;                                                      // count=1
  g0[1] = (unsigned)__builtin_amdgcn_readfirstlane((int)lds_off);  // lds_addr
  g0[2] = (unsigned)__builtin_amdgcn_readfirstlane((int)(ga & 0xFFFFFFFFull));
  g0[3] = ((unsigned)__builtin_amdgcn_readfirstlane(
               (int)((ga >> 32) & 0x01FFFFFFull))) |
          (2u << 30);                                              // type=2
  i32x8_t g1;
  g1[0] = 0x00010000;                       // data_size = 2 bytes
  g1[1] = 0x04000000;                       // tensor_dim0 = 1024 (bits 63:48)
  g1[2] = (int)(((unsigned)Nrows & 0xFFFFu) << 16);   // tensor_dim1 lo16
  g1[3] = 0x00200000 | (int)(((unsigned)Nrows >> 16) & 0xFFFFu); // tile_dim0=32
  g1[4] = 64;                               // tile_dim1 = 64 rows
  g1[5] = 1024;                             // tensor_dim0_stride = 1024 elems
  g1[6] = 0;
  g1[7] = 0;
  i32x4_t z4 = {};
#if defined(__clang_major__) && (__clang_major__ >= 23)
  i32x8_t z8 = {};
  __builtin_amdgcn_tensor_load_to_lds(g0, g1, z4, z4, z8, 0);
#else
  __builtin_amdgcn_tensor_load_to_lds(g0, g1, z4, z4, 0);
#endif
#else
  (void)gsrc; (void)ldsdst; (void)Nrows;
#endif
}

// ---- WMMA wrapper: D = A(16x32 f16) * B(32x16 f16) + C(16x16 f32) ----------
static __device__ __forceinline__ v8f wmma_f16(v16h a, v16h b, v8f c) {
  return __builtin_amdgcn_wmma_f32_16x16x32_f16(
      /*neg_a=*/false, a, /*neg_b=*/false, b,
      /*c_mod=*/(short)0, c, /*reuse_a=*/false, /*reuse_b=*/false);
}

// ---- A fragment (16x32, row-major source, leading dim = ld halves) ---------
static __device__ __forceinline__ v16h load_afrag(const _Float16* base, int ld) {
  int lane = threadIdx.x & 31;
  const _Float16* p = base + (lane & 15) * ld + ((lane >> 4) << 3);
  v8h lo = *(const v8h*)(p);        // K = k0 .. k0+7
  v8h hi = *(const v8h*)(p + 16);   // K = k0+16 .. k0+23
  v16h r;
#pragma unroll
  for (int i = 0; i < 8; ++i) { r[i] = lo[i]; r[i + 8] = hi[i]; }
  return r;
}

// ---- B fragment (32x16). Source stored TRANSPOSED: [N][K] row-major --------
static __device__ __forceinline__ v16h load_bfrag(const _Float16* baseT, int ld) {
  int lane = threadIdx.x & 31;
  const _Float16* p = baseT + (lane & 15) * ld + ((lane >> 4) << 3);
  v8h lo = *(const v8h*)(p);
  v8h hi = *(const v8h*)(p + 16);
  v16h r;
#pragma unroll
  for (int i = 0; i < 8; ++i) { r[i] = lo[i]; r[i + 8] = hi[i]; }
  return r;
}

// ---------------------------------------------------------------------------
// Kernel 1a: cast x (f32) -> f16
// ---------------------------------------------------------------------------
__global__ void cvt_f16_kernel(const float* __restrict__ in,
                               _Float16* __restrict__ out, int n) {
  int i = blockIdx.x * 256 + threadIdx.x;
  if (i < n) out[i] = (_Float16)in[i];
}

// ---------------------------------------------------------------------------
// Kernel 1b: transpose+cast weight W[K][N] (f32) -> Wt[N][K] (f16)
// ---------------------------------------------------------------------------
__global__ void transpose_w_kernel(const float* __restrict__ w,
                                   _Float16* __restrict__ wt, int K, int N) {
  int i = blockIdx.x * 256 + threadIdx.x;
  if (i < K * N) {
    int n = i / K;
    int k = i - n * K;
    wt[i] = (_Float16)w[(size_t)k * N + n];
  }
}

// ---------------------------------------------------------------------------
// Kernel 2: QKV GEMM.  Xh[BT][C] * WqkvT[3C][C]^T + bias -> scatter to
//           Q[B,H,T,D], K[B,H,T,D], Vt[B,H,D,T] (all f16).
// 4 waves/block (2x2), wave tile 64x64. Weight B-tiles streamed to LDS by TDM
// with TENSORcnt ping-pong; A fragments direct global b128 loads.
// ---------------------------------------------------------------------------
__global__ __launch_bounds__(128, 1)
void qkv_gemm_kernel(const _Float16* __restrict__ Xh,
                     const _Float16* __restrict__ Wt,
                     const float* __restrict__ bias,
                     _Float16* __restrict__ Qh,
                     _Float16* __restrict__ Kh,
                     _Float16* __restrict__ Vt) {
#if HAVE_TDM
  __shared__ _Float16 bstage[4][2][64 * 32];   // per-wave double buffer (32 KB)
#endif
  int wid = threadIdx.x >> 5;
  int m0 = blockIdx.x * 128 + (wid >> 1) * 64;
  int n0 = blockIdx.y * 128 + (wid & 1) * 64;

#if HAVE_TDM
  tdm_issue_64x32(Wt + (size_t)n0 * C_, &bstage[wid][0][0], NQKV_);
#endif

  v8f acc[4][4] = {};
  for (int kk = 0; kk < C_; kk += 32) {
    int cur = (kk >> 5) & 1;
#if HAVE_TDM
    if (kk + 32 < C_)
      tdm_issue_64x32(Wt + (size_t)n0 * C_ + kk + 32,
                      &bstage[wid][cur ^ 1][0], NQKV_);
#endif
    __builtin_prefetch(Xh + (size_t)(m0 + (threadIdx.x & 31)) * C_ + kk + 32,
                       0, 1);
    v16h a[4], b[4];
#pragma unroll
    for (int i = 0; i < 4; ++i)
      a[i] = load_afrag(Xh + (size_t)(m0 + 16 * i) * C_ + kk, C_);
#if HAVE_TDM
    if (kk + 32 < C_) wait_tensorcnt1(); else wait_tensorcnt0();
    asm volatile("" ::: "memory");
#pragma unroll
    for (int j = 0; j < 4; ++j)
      b[j] = load_bfrag(&bstage[wid][cur][16 * j * 32], 32);
#else
#pragma unroll
    for (int j = 0; j < 4; ++j)
      b[j] = load_bfrag(Wt + (size_t)(n0 + 16 * j) * C_ + kk, C_);
#endif
#pragma unroll
    for (int i = 0; i < 4; ++i)
#pragma unroll
      for (int j = 0; j < 4; ++j)
        acc[i][j] = wmma_f16(a[i], b[j], acc[i][j]);
  }

  int lane = threadIdx.x & 31;
  int cl = lane & 15;
  int rh = (lane >> 4) * 8;
#pragma unroll
  for (int i = 0; i < 4; ++i) {
#pragma unroll
    for (int j = 0; j < 4; ++j) {
      int col = n0 + 16 * j + cl;          // 0..3071
      int sec = col >> 10;                 // 0=Q 1=K 2=V
      int f   = col & (C_ - 1);
      int h   = f >> 6;
      int d   = f & 63;
      float bv = bias[col];
#pragma unroll
      for (int r = 0; r < 8; ++r) {
        int m  = m0 + 16 * i + rh + r;     // 0..4095
        int bb = m >> 11;                  // / T_
        int t  = m & (T_ - 1);
        _Float16 hv = (_Float16)(acc[i][j][r] + bv);
        if (sec == 0)
          Qh[((size_t)(bb * H_ + h) * T_ + t) * D_ + d] = hv;
        else if (sec == 1)
          Kh[((size_t)(bb * H_ + h) * T_ + t) * D_ + d] = hv;
        else
          Vt[((size_t)(bb * H_ + h) * D_ + d) * T_ + t] = hv;
      }
    }
  }
}

// ---------------------------------------------------------------------------
// Kernel 3: causal flash attention. One wave handles 32 q-rows x full D=64.
// Block: 128 threads = 4 waves (independent q strips). Online softmax in f32.
// ---------------------------------------------------------------------------
__global__ __launch_bounds__(128, 1)
void attn_kernel(const _Float16* __restrict__ Qh,
                 const _Float16* __restrict__ Kh,
                 const _Float16* __restrict__ Vt,
                 _Float16* __restrict__ Y) {
  __shared__ _Float16 plds[4][2][16 * 32];   // per-wave P staging tiles

  int wid  = threadIdx.x >> 5;
  int lane = threadIdx.x & 31;
  int strip = blockIdx.x * 4 + wid;          // 0..2047
  int bh = strip >> 6;                       // (b*H + h)
  int qs = (strip & 63) * 32;                // q-tile base row

  const _Float16* Qb = Qh + (size_t)bh * T_ * D_;
  const _Float16* Kb = Kh + (size_t)bh * T_ * D_;
  const _Float16* Vb = Vt + (size_t)bh * D_ * T_;

  int cl = lane & 15;
  int rh = (lane >> 4) * 8;

  // Q fragments resident: 2 m-tiles x 2 k-steps (K = D = 64)
  v16h qf[2][2];
#pragma unroll
  for (int i = 0; i < 2; ++i)
#pragma unroll
    for (int ks = 0; ks < 2; ++ks)
      qf[i][ks] = load_afrag(Qb + (size_t)(qs + 16 * i) * D_ + 32 * ks, D_);

  v8f O[2][4] = {};
  float rm[2][8], rl[2][8];
#pragma unroll
  for (int i = 0; i < 2; ++i)
#pragma unroll
    for (int r = 0; r < 8; ++r) { rm[i][r] = NEG_INF; rl[i][r] = 0.0f; }

  const float scale = 0.125f;                // 1/sqrt(64)
  int nsteps = (qs >> 5) + 1;                // causal: keys up to qs+31

  for (int s = 0; s < nsteps; ++s) {
    int kb = s * 32;
    v16h kf[2][2], vf[4];
#pragma unroll
    for (int jn = 0; jn < 2; ++jn)
#pragma unroll
      for (int ks = 0; ks < 2; ++ks)
        kf[jn][ks] = load_bfrag(Kb + (size_t)(kb + 16 * jn) * D_ + 32 * ks, D_);
#pragma unroll
    for (int jd = 0; jd < 4; ++jd)
      vf[jd] = load_bfrag(Vb + (size_t)(16 * jd) * T_ + kb, T_);

#pragma unroll
    for (int i = 0; i < 2; ++i) {
      // S = Q * K^T for 32 keys (two 16-col accumulators)
      v8f S0 = {}, S1 = {};
      S0 = wmma_f16(qf[i][0], kf[0][0], S0);
      S0 = wmma_f16(qf[i][1], kf[0][1], S0);
      S1 = wmma_f16(qf[i][0], kf[1][0], S1);
      S1 = wmma_f16(qf[i][1], kf[1][1], S1);

      float s0[8], s1[8];
#pragma unroll
      for (int r = 0; r < 8; ++r) { s0[r] = S0[r] * scale; s1[r] = S1[r] * scale; }

      if (kb + 31 > qs + 16 * i) {           // causal mask needed
        int qrow = qs + 16 * i + rh;
#pragma unroll
        for (int r = 0; r < 8; ++r) {
          if (kb + cl      > qrow + r) s0[r] = NEG_INF;
          if (kb + 16 + cl > qrow + r) s1[r] = NEG_INF;
        }
      }

      // row max across the 16 lanes of each half-wave
      float mx[8];
#pragma unroll
      for (int r = 0; r < 8; ++r) mx[r] = fmaxf(s0[r], s1[r]);
#pragma unroll
      for (int off = 1; off < 16; off <<= 1)
#pragma unroll
        for (int r = 0; r < 8; ++r)
          mx[r] = fmaxf(mx[r], __shfl_xor(mx[r], off, 32));

      float corr[8], ps[8];
#pragma unroll
      for (int r = 0; r < 8; ++r) {
        float mn = fmaxf(rm[i][r], mx[r]);
        corr[r]  = __expf(rm[i][r] - mn);
        rm[i][r] = mn;
        float p0 = __expf(s0[r] - mn);
        float p1 = __expf(s1[r] - mn);
        ps[r] = p0 + p1;
        plds[wid][i][(rh + r) * 32 + cl]      = (_Float16)p0;
        plds[wid][i][(rh + r) * 32 + cl + 16] = (_Float16)p1;
      }
#pragma unroll
      for (int off = 1; off < 16; off <<= 1)
#pragma unroll
        for (int r = 0; r < 8; ++r)
          ps[r] += __shfl_xor(ps[r], off, 32);
#pragma unroll
      for (int r = 0; r < 8; ++r)
        rl[i][r] = rl[i][r] * corr[r] + ps[r];

      // rescale O, then O += P * V  (P re-read as A-fragment via LDS;
      // DS ops are in-order within a wave, no barrier needed)
#pragma unroll
      for (int jd = 0; jd < 4; ++jd)
#pragma unroll
        for (int r = 0; r < 8; ++r)
          O[i][jd][r] *= corr[r];

      v16h af = load_afrag(&plds[wid][i][0], 32);
#pragma unroll
      for (int jd = 0; jd < 4; ++jd)
        O[i][jd] = wmma_f16(af, vf[jd], O[i][jd]);
    }
  }

  // normalize and write Y[B,T,C] as f16
  int b = bh >> 4;
  int h = bh & 15;
#pragma unroll
  for (int i = 0; i < 2; ++i)
#pragma unroll
    for (int jd = 0; jd < 4; ++jd)
#pragma unroll
      for (int r = 0; r < 8; ++r) {
        int t = qs + 16 * i + rh + r;
        Y[((size_t)(b * T_ + t)) * C_ + h * 64 + 16 * jd + cl] =
            (_Float16)(O[i][jd][r] / rl[i][r]);
      }
}

// ---------------------------------------------------------------------------
// Kernel 4: output projection.  Y[BT][C] * WprojT[C][C]^T + bias -> out (f32)
// Same TDM-staged structure as the QKV GEMM.
// ---------------------------------------------------------------------------
__global__ __launch_bounds__(128, 1)
void proj_gemm_kernel(const _Float16* __restrict__ Yh,
                      const _Float16* __restrict__ Wt,
                      const float* __restrict__ bias,
                      float* __restrict__ out) {
#if HAVE_TDM
  __shared__ _Float16 bstage[4][2][64 * 32];
#endif
  int wid = threadIdx.x >> 5;
  int m0 = blockIdx.x * 128 + (wid >> 1) * 64;
  int n0 = blockIdx.y * 128 + (wid & 1) * 64;

#if HAVE_TDM
  tdm_issue_64x32(Wt + (size_t)n0 * C_, &bstage[wid][0][0], C_);
#endif

  v8f acc[4][4] = {};
  for (int kk = 0; kk < C_; kk += 32) {
    int cur = (kk >> 5) & 1;
#if HAVE_TDM
    if (kk + 32 < C_)
      tdm_issue_64x32(Wt + (size_t)n0 * C_ + kk + 32,
                      &bstage[wid][cur ^ 1][0], C_);
#endif
    __builtin_prefetch(Yh + (size_t)(m0 + (threadIdx.x & 31)) * C_ + kk + 32,
                       0, 1);
    v16h a[4], b[4];
#pragma unroll
    for (int i = 0; i < 4; ++i)
      a[i] = load_afrag(Yh + (size_t)(m0 + 16 * i) * C_ + kk, C_);
#if HAVE_TDM
    if (kk + 32 < C_) wait_tensorcnt1(); else wait_tensorcnt0();
    asm volatile("" ::: "memory");
#pragma unroll
    for (int j = 0; j < 4; ++j)
      b[j] = load_bfrag(&bstage[wid][cur][16 * j * 32], 32);
#else
#pragma unroll
    for (int j = 0; j < 4; ++j)
      b[j] = load_bfrag(Wt + (size_t)(n0 + 16 * j) * C_ + kk, C_);
#endif
#pragma unroll
    for (int i = 0; i < 4; ++i)
#pragma unroll
      for (int j = 0; j < 4; ++j)
        acc[i][j] = wmma_f16(a[i], b[j], acc[i][j]);
  }

  int lane = threadIdx.x & 31;
  int cl = lane & 15;
  int rh = (lane >> 4) * 8;
#pragma unroll
  for (int i = 0; i < 4; ++i)
#pragma unroll
    for (int j = 0; j < 4; ++j) {
      int col = n0 + 16 * j + cl;
      float bv = bias[col];
#pragma unroll
      for (int r = 0; r < 8; ++r) {
        int m = m0 + 16 * i + rh + r;
        out[(size_t)m * C_ + col] = acc[i][j][r] + bv;
      }
    }
}

// ---------------------------------------------------------------------------
// Host launcher
// ---------------------------------------------------------------------------
extern "C" void kernel_launch(void* const* d_in, const int* in_sizes, int n_in,
                              void* d_out, int out_size, void* d_ws, size_t ws_size,
                              hipStream_t stream) {
  const float* x        = (const float*)d_in[0];
  const float* c_attn_w = (const float*)d_in[1];
  const float* c_attn_b = (const float*)d_in[2];
  const float* c_proj_w = (const float*)d_in[3];
  const float* c_proj_b = (const float*)d_in[4];

  char* ws = (char*)d_ws;
  size_t off = 0;
  auto carve = [&](size_t bytes) -> char* {
    char* p = ws + off;
    off += (bytes + 255) & ~(size_t)255;
    return p;
  };

  _Float16* Xh     = (_Float16*)carve((size_t)BT_ * C_ * 2);          // 8 MiB
  _Float16* Wqkvt  = (_Float16*)carve((size_t)NQKV_ * C_ * 2);        // 6 MiB
  _Float16* Wprojt = (_Float16*)carve((size_t)C_ * C_ * 2);           // 2 MiB
  _Float16* Qh     = (_Float16*)carve((size_t)B_ * H_ * T_ * D_ * 2); // 8 MiB
  _Float16* Kh     = (_Float16*)carve((size_t)B_ * H_ * T_ * D_ * 2); // 8 MiB
  _Float16* Vt     = (_Float16*)carve((size_t)B_ * H_ * D_ * T_ * 2); // 8 MiB
  _Float16* Yh     = (_Float16*)carve((size_t)BT_ * C_ * 2);          // 8 MiB
  (void)ws_size; (void)in_sizes; (void)n_in; (void)out_size;

  // 1. precision conversion + weight transposes
  int nX = BT_ * C_;
  cvt_f16_kernel<<<(nX + 255) / 256, 256, 0, stream>>>(x, Xh, nX);
  int nWq = C_ * NQKV_;
  transpose_w_kernel<<<(nWq + 255) / 256, 256, 0, stream>>>(c_attn_w, Wqkvt, C_, NQKV_);
  int nWp = C_ * C_;
  transpose_w_kernel<<<(nWp + 255) / 256, 256, 0, stream>>>(c_proj_w, Wprojt, C_, C_);

  // 2. QKV projection (4096 x 3072 x 1024)
  qkv_gemm_kernel<<<dim3(BT_ / 128, NQKV_ / 128), 128, 0, stream>>>(
      Xh, Wqkvt, c_attn_b, Qh, Kh, Vt);

  // 3. causal flash attention (2048 q-strips, 4 waves per block)
  attn_kernel<<<(B_ * H_ * (T_ / 32)) / 4, 128, 0, stream>>>(Qh, Kh, Vt, Yh);

  // 4. output projection (4096 x 1024 x 1024) -> f32 out
  proj_gemm_kernel<<<dim3(BT_ / 128, C_ / 128), 128, 0, stream>>>(
      Yh, Wprojt, c_proj_b, (float*)d_out);
}
